// TimeMoERouter_3435973837292
// MI455X (gfx1250) — matrix operand order repaired and checked by reading, not compiled, not run on hardware
//
#include <hip/hip_runtime.h>

// ---------------------------------------------------------------------------
// TimeMoE router for MI455X (gfx1250, wave32).
// GEMMs via v_wmma_f32_16x16x32_f16 (fp32 accum). Weights converted once per
// launch to f16 in d_ws; activations staged fp32->f16 through LDS with
// v_cvt_pkrtz_f16_f32. 400MB dispatch/combine written with non-temporal
// stores to keep the 192MB L2 for reused activations.
// ---------------------------------------------------------------------------

#define NT   4096      // B*S tokens
#define SS   512
#define BBAT 8
#define HH   1024
#define EE   8
#define CAPS 1536
#define NH   8
#define DH   128
#define KPAD 1056      // 1026 padded up to a multiple of 32

typedef __attribute__((ext_vector_type(16))) _Float16 v16h;
typedef __attribute__((ext_vector_type(2)))  __fp16   v2fp16;  // cvt_pkrtz result type
typedef __attribute__((ext_vector_type(8)))  float    v8f;
typedef __attribute__((ext_vector_type(4)))  float    vf4;     // native vec for NT stores

union FragH { v16h v; uint4 q[2]; };
union FragF { v8f  v; float f[8]; };

__device__ __forceinline__ unsigned short f2h_bits(float f) {
  union { _Float16 h; unsigned short u; } c; c.h = (_Float16)f; return c.u;
}
// single v_cvt_pkrtz_f16_f32: two f32 -> packed 2xf16 in one VGPR
__device__ __forceinline__ unsigned int pack2h(float a, float b) {
  union { v2fp16 h; unsigned int u; } c;
  c.h = __builtin_amdgcn_cvt_pkrtz(a, b);
  return c.u;
}

// --------------------------- weight conversion -----------------------------
__global__ void k_f32_to_f16(const float* __restrict__ s,
                             unsigned short* __restrict__ d, int n) {
  for (int i = blockIdx.x * blockDim.x + threadIdx.x; i < n;
       i += gridDim.x * blockDim.x)
    d[i] = f2h_bits(__builtin_nontemporal_load(s + i));
}

// pad-convert: src [N,Ks] f32 -> dst [N,Kd] f16 (zero pad K)
__global__ void k_pad_f16(const float* __restrict__ s,
                          unsigned short* __restrict__ d,
                          int N, int Ks, int Kd) {
  int tot = N * Kd;
  for (int i = blockIdx.x * blockDim.x + threadIdx.x; i < tot;
       i += gridDim.x * blockDim.x) {
    int r = i / Kd, k = i - r * Kd;
    d[i] = (k < Ks) ? f2h_bits(__builtin_nontemporal_load(s + (size_t)r * Ks + k))
                    : (unsigned short)0;
  }
}

// --------------------------- elementwise front-end -------------------------
__global__ void k_combined(const float* __restrict__ hidden,
                           float* __restrict__ comb) {
  const float w = 0.26179938779914946f;       // 2*pi/24
  int tot = NT * KPAD;
  for (int i = blockIdx.x * blockDim.x + threadIdx.x; i < tot;
       i += gridDim.x * blockDim.x) {
    int row = i / KPAD, col = i - row * KPAD;
    int s = row & (SS - 1);
    float v;
    if (col < HH)            v = hidden[(size_t)row * HH + col];
    else if (col == HH)      v = (float)s;
    else if (col == HH + 1)  v = sinf((float)s * w);
    else                     v = 0.f;
    comb[i] = v;
  }
}

__global__ void k_se1(const float* __restrict__ w1, const float* __restrict__ b1,
                      float* __restrict__ out) {
  const float w = 0.26179938779914946f;
  int tot = NT * 256;
  for (int i = blockIdx.x * blockDim.x + threadIdx.x; i < tot;
       i += gridDim.x * blockDim.x) {
    int row = i >> 8, j = i & 255;
    float se = sinf((float)(row & (SS - 1)) * w);
    out[i] = fmaxf(se * w1[j] + b1[j], 0.f);
  }
}

// split qkv [4096,3072] -> q f32 [64][512][128], k f16 [64][512][128],
// vT f16 [64][128][512]
__global__ void k_qkv_split(const float* __restrict__ qkv,
                            float* __restrict__ qf,
                            unsigned short* __restrict__ kh,
                            unsigned short* __restrict__ vT) {
  int tot = NT * HH;
  for (int i = blockIdx.x * blockDim.x + threadIdx.x; i < tot;
       i += gridDim.x * blockDim.x) {
    int d = i & (DH - 1);
    int s = (i >> 7) & (SS - 1);
    int z = i >> 16;                 // b*8+h
    int b = z >> 3, h = z & 7;
    size_t src = ((size_t)(b * SS + s)) * (3 * HH) + h * DH + d;
    qf[i] = qkv[src];
    kh[i] = f2h_bits(qkv[src + HH]);
    vT[(size_t)z * DH * SS + (size_t)d * SS + s] = f2h_bits(qkv[src + 2 * HH]);
  }
}

// --------------------------- generic WMMA GEMM -----------------------------
// C[M,N] = act( A[M,K](f32) * W[N,K]^T(f16) + bias + addF + addB ), batched.
// BM=128, BN=64, BK=32. 256 threads = 8 waves; wave w owns rows [16w,16w+16).
__global__ void __launch_bounds__(256) k_wmma_gemm(
    const float* __restrict__ A, long long strA, int lda,
    const unsigned short* __restrict__ W, long long strW,
    const float* __restrict__ bias,
    const float* __restrict__ addF,                 // residual, C-shaped
    const float* __restrict__ addB, int addBmod,    // broadcast over row%mod
    float* __restrict__ C, int ldc, long long cs0, long long cs1, int hz,
    int M, int N, int K, int relu) {
  __shared__ __align__(16) unsigned short As[128 * 32];
  __shared__ __align__(16) unsigned short Bs[64 * 32];

  const int tid = threadIdx.x;
  const int n0 = blockIdx.x * 64;
  const int m0 = blockIdx.y * 128;
  const int z  = blockIdx.z;

  const float* Ab = A + (size_t)z * strA;
  const unsigned short* Wb = W + (size_t)z * strW;
  const long long coff = (long long)(z / hz) * cs0 + (long long)(z % hz) * cs1;
  float* Cb = C + coff;
  const float* AFb = addF ? (addF + coff) : (const float*)0;

  const int wave = tid >> 5, lane = tid & 31;
  const int lrow = lane & 15, hi16 = lane >> 4;
  const int mB = wave * 16;

  FragF acc[4];
#pragma unroll
  for (int j = 0; j < 4; ++j)
#pragma unroll
    for (int i = 0; i < 8; ++i) acc[j].f[i] = 0.f;

  const int ra = tid >> 1, ca = (tid & 1) * 16;   // A stage: 16 floats/thread
  const int rb = tid >> 2, cb = (tid & 3) * 8;    // B stage: 8 halves/thread

  for (int k0 = 0; k0 < K; k0 += 32) {
    __syncthreads();
    {   // stage A tile (f32 -> f16 into LDS via v_cvt_pkrtz_f16_f32)
      const float4* ap = (const float4*)(Ab + (size_t)(m0 + ra) * lda + k0 + ca);
      float4 f0 = ap[0], f1 = ap[1], f2 = ap[2], f3 = ap[3];
      uint4 lo = make_uint4(pack2h(f0.x, f0.y), pack2h(f0.z, f0.w),
                            pack2h(f1.x, f1.y), pack2h(f1.z, f1.w));
      uint4 hi = make_uint4(pack2h(f2.x, f2.y), pack2h(f2.z, f2.w),
                            pack2h(f3.x, f3.y), pack2h(f3.z, f3.w));
      uint4* d = (uint4*)&As[ra * 32 + ca];
      d[0] = lo; d[1] = hi;
    }
    {   // stage B tile (f16 weights; zero rows past N)
      int n = n0 + rb;
      uint4 v = make_uint4(0u, 0u, 0u, 0u);
      if (n < N) v = *(const uint4*)(Wb + (size_t)n * K + k0 + cb);
      *(uint4*)&Bs[rb * 32 + cb] = v;
    }
    if (k0 + 32 < K) {   // global_prefetch_b8 of the next K-tile
      __builtin_prefetch(Ab + (size_t)(m0 + ra) * lda + (k0 + 32) + ca, 0, 1);
      if (n0 + rb < N)
        __builtin_prefetch(Wb + (size_t)(n0 + rb) * K + (k0 + 32) + cb, 0, 1);
    }
    __syncthreads();

    // A fragment: lane<16 holds K0..15, lane>=16 holds K16..31
    FragH a;
    a.q[0] = *(const uint4*)&As[(mB + lrow) * 32 + hi16 * 16];
    a.q[1] = *(const uint4*)&As[(mB + lrow) * 32 + hi16 * 16 + 8];
#pragma unroll
    for (int j = 0; j < 4; ++j) {
      // B fragment: lane<16 col n, K0..15; lane>=16 col n, K16..31
      FragH b;
      const unsigned short* bp = &Bs[(j * 16 + lrow) * 32 + hi16 * 16];
      b.q[0] = *(const uint4*)bp;
      b.q[1] = *(const uint4*)(bp + 8);
      acc[j].v = __builtin_amdgcn_wmma_f32_16x16x32_f16(
          false, a.v, false, b.v, (short)0, acc[j].v, false, false);
    }
  }

  // epilogue: C/D layout — VGPR r: M=r (lanes 0-15) / M=r+8 (lanes 16-31)
#pragma unroll
  for (int j = 0; j < 4; ++j) {
    int n = n0 + j * 16 + lrow;
    if (n >= N) continue;
    float bv = bias ? bias[n] : 0.f;
#pragma unroll
    for (int r = 0; r < 8; ++r) {
      int row = m0 + mB + r + hi16 * 8;
      float v = acc[j].f[r] + bv;
      size_t ci = (size_t)row * ldc + n;
      if (AFb)  v += AFb[ci];
      if (addB) v += addB[(size_t)(row % addBmod) * N + n];
      if (relu) v = fmaxf(v, 0.f);
      Cb[ci] = v;
    }
  }
}

// --------------------------- row softmax (in place) ------------------------
__global__ void k_softmax_rows(float* __restrict__ X, int ncols, float scale) {
  const size_t base = (size_t)blockIdx.x * ncols;
  int tid = threadIdx.x, lane = tid & 31, wave = tid >> 5;
  __shared__ float red[8];
  float m = -1e30f;
  for (int c = tid; c < ncols; c += blockDim.x)
    m = fmaxf(m, X[base + c] * scale);
  for (int o = 16; o; o >>= 1) m = fmaxf(m, __shfl_xor(m, o, 32));
  if (lane == 0) red[wave] = m;
  __syncthreads();
  if (tid == 0) {
    float t = red[0];
    for (int i = 1; i < 8; ++i) t = fmaxf(t, red[i]);
    red[0] = t;
  }
  __syncthreads();
  m = red[0];
  __syncthreads();
  float s = 0.f;
  for (int c = tid; c < ncols; c += blockDim.x)
    s += __expf(X[base + c] * scale - m);
  for (int o = 16; o; o >>= 1) s += __shfl_xor(s, o, 32);
  if (lane == 0) red[wave] = s;
  __syncthreads();
  if (tid == 0) {
    float t = 0.f;
    for (int i = 0; i < 8; ++i) t += red[i];
    red[0] = t;
  }
  __syncthreads();
  float inv = 1.0f / red[0];
  for (int c = tid; c < ncols; c += blockDim.x)
    X[base + c] = __expf(X[base + c] * scale - m) * inv;
}

// --------------------------- output fill + router --------------------------
// dispatch/combine are write-only, never re-read on device: use NT stores
// so 400MB of zeros doesn't rinse L2.
__global__ void k_zero4(float* __restrict__ p, long long n4) {
  vf4 z = {0.f, 0.f, 0.f, 0.f};
  vf4* p4 = (vf4*)p;
  for (long long i = (long long)blockIdx.x * blockDim.x + threadIdx.x; i < n4;
       i += (long long)gridDim.x * blockDim.x)
    __builtin_nontemporal_store(z, p4 + i);
}
__global__ void k_zero(float* __restrict__ p, int n) {
  int i = blockIdx.x * blockDim.x + threadIdx.x;
  if (i < n) p[i] = 0.f;
}

__global__ void k_router(const float* __restrict__ logits,
                         float* __restrict__ out,
                         float* __restrict__ probsum) {
  int t = blockIdx.x * blockDim.x + threadIdx.x;
  if (t >= NT) return;
  float p[EE];
  float m = -1e30f;
#pragma unroll
  for (int e = 0; e < EE; ++e) { p[e] = logits[(size_t)t * EE + e]; m = fmaxf(m, p[e]); }
  float s = 0.f;
#pragma unroll
  for (int e = 0; e < EE; ++e) { p[e] = __expf(p[e] - m); s += p[e]; }
  float inv = 1.0f / s;
#pragma unroll
  for (int e = 0; e < EE; ++e) p[e] *= inv;

  const size_t D1 = (size_t)NT * EE * CAPS;
  float* probs_out = out + 2 * D1;
#pragma unroll
  for (int e = 0; e < EE; ++e) {
    probs_out[(size_t)t * EE + e] = p[e];
    atomicAdd(&probsum[e], p[e]);
  }
  int i1 = 0;
#pragma unroll
  for (int e = 1; e < EE; ++e) if (p[e] > p[i1]) i1 = e;
  int i2 = (i1 == 0) ? 1 : 0;
#pragma unroll
  for (int e = 0; e < EE; ++e) if (e != i1 && p[e] > p[i2]) i2 = e;
  float denom = p[i1] + p[i2];
  __builtin_nontemporal_store(1.f, out + ((size_t)t * EE + i1) * CAPS);
  __builtin_nontemporal_store(1.f, out + ((size_t)t * EE + i2) * CAPS);
  __builtin_nontemporal_store(p[i1] / denom, out + D1 + ((size_t)t * EE + i1) * CAPS);
  __builtin_nontemporal_store(p[i2] / denom, out + D1 + ((size_t)t * EE + i2) * CAPS);
}

__global__ void k_aux(const float* __restrict__ probsum, float* __restrict__ out) {
  if (blockIdx.x == 0 && threadIdx.x == 0) {
    const size_t D1 = (size_t)NT * EE * CAPS;
    float a = 0.f;
    for (int e = 0; e < EE; ++e) {
      float pm = probsum[e] / (float)NT;
      a += pm * logf(pm * (float)EE + 1e-9f);
    }
    out[2 * D1 + (size_t)NT * EE] = a;
  }
}

// ---------------------------------------------------------------------------
extern "C" void kernel_launch(void* const* d_in, const int* in_sizes, int n_in,
                              void* d_out, int out_size, void* d_ws, size_t ws_size,
                              hipStream_t stream) {
  (void)in_sizes; (void)n_in; (void)out_size; (void)ws_size;
  const float* hidden     = (const float*)d_in[0];
  const float* pos_emb    = (const float*)d_in[1];
  const float* se_w1      = (const float*)d_in[2];
  const float* se_b1      = (const float*)d_in[3];
  const float* se_w2      = (const float*)d_in[4];
  const float* se_b2      = (const float*)d_in[5];
  const float* sexp_w     = (const float*)d_in[6];
  const float* sexp_b     = (const float*)d_in[7];
  const float* te_w1      = (const float*)d_in[8];
  const float* te_b1      = (const float*)d_in[9];
  const float* te_w2      = (const float*)d_in[10];
  const float* te_b2      = (const float*)d_in[11];
  const float* attn_in_w  = (const float*)d_in[12];
  const float* attn_in_b  = (const float*)d_in[13];
  const float* attn_out_w = (const float*)d_in[14];
  const float* attn_out_b = (const float*)d_in[15];
  const float* tr_w1      = (const float*)d_in[16];
  const float* tr_b1      = (const float*)d_in[17];
  const float* tr_w2      = (const float*)d_in[18];
  const float* tr_b2      = (const float*)d_in[19];
  float* out = (float*)d_out;

  // ----- scratch carve-out -----
  char* base = (char*)d_ws;
  size_t cur = 0;
  auto alloc = [&](size_t bytes) -> void* {
    void* p = base + cur;
    cur = (cur + bytes + 255) & ~(size_t)255;
    return p;
  };
  // f16 weight copies
  unsigned short* w_te1  = (unsigned short*)alloc((size_t)HH * KPAD * 2);
  unsigned short* w_te2  = (unsigned short*)alloc((size_t)HH * HH * 2);
  unsigned short* w_ain  = (unsigned short*)alloc((size_t)3 * HH * HH * 2);
  unsigned short* w_aout = (unsigned short*)alloc((size_t)HH * HH * 2);
  unsigned short* w_tr1  = (unsigned short*)alloc((size_t)HH * HH * 2);
  unsigned short* w_sexp = (unsigned short*)alloc((size_t)HH * 256 * 2);
  unsigned short* w_se2  = (unsigned short*)alloc((size_t)256 * 256 * 2);
  unsigned short* w_tr2  = (unsigned short*)alloc((size_t)EE * HH * 2);
  unsigned short* kh     = (unsigned short*)alloc((size_t)NT * HH * 2);
  unsigned short* vT     = (unsigned short*)alloc((size_t)NT * HH * 2);
  // fp32 activations
  float* combined = (float*)alloc((size_t)NT * KPAD * 4);
  float* seas1    = (float*)alloc((size_t)NT * 256 * 4);
  float* seas2    = (float*)alloc((size_t)NT * 256 * 4);
  float* seasonal = (float*)alloc((size_t)NT * HH * 4);
  float* enc1     = (float*)alloc((size_t)NT * HH * 4);
  float* enc      = (float*)alloc((size_t)NT * HH * 4);
  float* qkv      = (float*)alloc((size_t)NT * 3 * HH * 4);
  float* qf       = (float*)alloc((size_t)NT * HH * 4);
  float* scores   = (float*)alloc((size_t)BBAT * NH * SS * SS * 4); // softmax in place
  float* concat   = (float*)alloc((size_t)NT * HH * 4);
  float* encA     = (float*)alloc((size_t)NT * HH * 4);
  float* h1       = (float*)alloc((size_t)NT * HH * 4);
  float* logits   = (float*)alloc((size_t)NT * EE * 4);
  float* probsum  = (float*)alloc(64 * 4);

  auto cvt = [&](const float* s, unsigned short* d, int n) {
    int blocks = (n + 256 * 4 - 1) / (256 * 4);
    if (blocks > 4096) blocks = 4096;
    k_f32_to_f16<<<blocks, 256, 0, stream>>>(s, d, n);
  };
  cvt(te_w2, w_te2, HH * HH);
  cvt(attn_in_w, w_ain, 3 * HH * HH);
  cvt(attn_out_w, w_aout, HH * HH);
  cvt(tr_w1, w_tr1, HH * HH);
  cvt(sexp_w, w_sexp, HH * 256);
  cvt(se_w2, w_se2, 256 * 256);
  cvt(tr_w2, w_tr2, EE * HH);
  k_pad_f16<<<4096, 256, 0, stream>>>(te_w1, w_te1, HH, HH + 2, KPAD);

  k_combined<<<4096, 256, 0, stream>>>(hidden, combined);
  k_se1<<<2048, 256, 0, stream>>>(se_w1, se_b1, seas1);

  auto gemm = [&](const float* A, long long strA, int lda,
                  const unsigned short* W, long long strW,
                  const float* bias, const float* addF,
                  const float* addB, int addBmod,
                  float* C, int ldc, long long cs0, long long cs1, int hz,
                  int M, int N, int K, int relu, int batch) {
    dim3 g((N + 63) / 64, M / 128, batch);
    k_wmma_gemm<<<g, 256, 0, stream>>>(A, strA, lda, W, strW, bias, addF,
                                       addB, addBmod, C, ldc, cs0, cs1, hz,
                                       M, N, K, relu);
  };

  // seasonal MLP
  gemm(seas1, 0, 256, w_se2, 0, se_b2, 0, 0, 1, seas2, 256, 0, 0, 1,
       NT, 256, 256, 0, 1);
  gemm(seas2, 0, 256, w_sexp, 0, sexp_b, 0, 0, 1, seasonal, HH, 0, 0, 1,
       NT, HH, 256, 0, 1);
  // time encoder
  gemm(combined, 0, KPAD, w_te1, 0, te_b1, 0, 0, 1, enc1, HH, 0, 0, 1,
       NT, HH, KPAD, 1, 1);
  gemm(enc1, 0, HH, w_te2, 0, te_b2, seasonal, pos_emb, SS, enc, HH, 0, 0, 1,
       NT, HH, HH, 0, 1);
  // attention
  gemm(enc, 0, HH, w_ain, 0, attn_in_b, 0, 0, 1, qkv, 3 * HH, 0, 0, 1,
       NT, 3 * HH, HH, 0, 1);
  k_qkv_split<<<4096, 256, 0, stream>>>(qkv, qf, kh, vT);
  // scores[z] = q[z] @ k[z]^T   (64 batches of [512,128]x[128,512])
  gemm(qf, (long long)SS * DH, DH, kh, (long long)SS * DH, 0, 0, 0, 1,
       scores, SS, (long long)SS * SS, 0, 1, SS, SS, DH, 0, BBAT * NH);
  k_softmax_rows<<<BBAT * NH * SS, 256, 0, stream>>>(scores, SS,
                                                     0.08838834764831845f);
  // o[z] = P[z] @ v[z], scattered into concat[b, s, h*128 + d]
  gemm(scores, (long long)SS * SS, SS, vT, (long long)DH * SS, 0, 0, 0, 1,
       concat, HH, (long long)SS * HH, DH, NH, SS, DH, SS, 0, BBAT * NH);
  gemm(concat, 0, HH, w_aout, 0, attn_out_b, 0, 0, 1, encA, HH, 0, 0, 1,
       NT, HH, HH, 0, 1);
  // router head
  gemm(encA, 0, HH, w_tr1, 0, tr_b1, 0, 0, 1, h1, HH, 0, 0, 1,
       NT, HH, HH, 1, 1);
  gemm(h1, 0, HH, w_tr2, 0, tr_b2, 0, 0, 1, logits, EE, 0, 0, 1,
       NT, EE, HH, 0, 1);

  // outputs: zero-fill dispatch+combine (streaming NT float4 stores), scatter,
  // probs, aux
  const long long D1 = (long long)NT * EE * CAPS;
  k_zero4<<<8192, 256, 0, stream>>>(out, (2 * D1) / 4);
  k_zero<<<1, 64, 0, stream>>>(probsum, 64);
  k_router<<<(NT + 255) / 256, 256, 0, stream>>>(logits, out, probsum);
  k_aux<<<1, 32, 0, stream>>>(probsum, out);
}
